// VolitionBrain_66898410602604
// MI455X (gfx1250) — compile-verified, matrix operand order checked by reading, not run on hardware
//
#include <hip/hip_runtime.h>
#include <hip/hip_bf16.h>
#include <math.h>

#define Dm 384
#define Bsz 32
#define Tlen 1024
#define VOC 256
#define GAINF 14.0f
#define TD3 1152   // 3*D
#define K2D 768    // 2*D

typedef __attribute__((ext_vector_type(16))) _Float16 v16h;
typedef __attribute__((ext_vector_type(8)))  float    v8f;

// ---------------- WMMA helpers ----------------
// A/B fragment layout (ISA 7.12.2, 16-bit, wave32):
// lane L (hsel = L>>4, l15 = L&15) element e -> k = 16*(e>>3) + 8*hsel + (e&7)
__device__ __forceinline__ v16h frag_h(const _Float16* rowp, int hsel) {
  v16h f;
  const _Float16* p0 = rowp + 8 * hsel;
  const _Float16* p1 = rowp + 16 + 8 * hsel;
#pragma unroll
  for (int j = 0; j < 8; ++j) { f[j] = p0[j]; f[j + 8] = p1[j]; }
  return f;
}
__device__ __forceinline__ v16h frag_f(const float* rowp, int hsel) {
  v16h f;
  const float* p0 = rowp + 8 * hsel;
  const float* p1 = rowp + 16 + 8 * hsel;
#pragma unroll
  for (int j = 0; j < 8; ++j) { f[j] = (_Float16)p0[j]; f[j + 8] = (_Float16)p1[j]; }
  return f;
}
__device__ __forceinline__ v8f wmma_acc(v16h a, v16h b, v8f c) {
  return __builtin_amdgcn_wmma_f32_16x16x32_f16(false, a, false, b, (short)0, c, false, false);
}

__device__ __forceinline__ float sigmf(float x) { return 1.f / (1.f + expf(-x)); }
__device__ __forceinline__ float geluf(float x) { return 0.5f * x * (1.f + erff(x * 0.70710678118f)); }

__device__ __forceinline__ unsigned hashu(unsigned a) {
  a ^= a >> 16; a *= 0x7feb352dU; a ^= a >> 15; a *= 0x846ca68bU; a ^= a >> 16; return a;
}
__device__ __forceinline__ float gnoise(unsigned seed) {
  unsigned h1 = hashu(seed * 2654435761u + 0x9e3779b9u);
  unsigned h2 = hashu(h1 + 0x85ebca6bu);
  float u1 = (float)(h1 & 0xFFFFFF) * (1.0f / 16777216.0f) + 1e-7f;
  float u2 = (float)(h2 & 0xFFFFFF) * (1.0f / 16777216.0f);
  return sqrtf(-2.f * logf(u1)) * cosf(6.28318530718f * u2);
}

// ---------------- K0: weight conversion to f16 ----------------
__global__ void k_prep(const float* fwih, const float* fwhh,
                       const float* swih, const float* swhh,
                       const float* bwih, const float* bwhh,
                       const float* trw, const float* wsw, const float* pfcw,
                       _Float16* WcF, _Float16* WcS, _Float16* WcB,
                       _Float16* TrH, _Float16* WsH, _Float16* PfcH) {
  long i0 = (long)blockIdx.x * blockDim.x + threadIdx.x;
  long stride = (long)gridDim.x * blockDim.x;
  for (long i = i0; i < (long)TD3 * K2D; i += stride) {
    int r = (int)(i / K2D), c = (int)(i % K2D);
    float vF = c < Dm ? fwih[r * Dm + c] : fwhh[r * Dm + (c - Dm)];
    float vS = c < Dm ? swih[r * Dm + c] : swhh[r * Dm + (c - Dm)];
    float vB = c < Dm ? bwih[r * Dm + c] : bwhh[r * Dm + (c - Dm)];
    WcF[i] = (_Float16)vF; WcS[i] = (_Float16)vS; WcB[i] = (_Float16)vB;
  }
  for (long i = i0; i < (long)Dm * Dm; i += stride) TrH[i] = (_Float16)trw[i];
  for (long i = i0; i < (long)Dm * K2D; i += stride) {
    WsH[i] = (_Float16)wsw[i];
    PfcH[i] = (_Float16)pfcw[i];
  }
}

__global__ void k_norm_soma(const float* soma, _Float16* swn) {
  int r = blockIdx.x * blockDim.x + threadIdx.x;
  if (r >= VOC) return;
  float s = 0.f;
  for (int k = 0; k < Dm; ++k) { float v = soma[r * Dm + k]; s += v * v; }
  float inv = 1.f / sqrtf(s);
  for (int k = 0; k < Dm; ++k) swn[r * Dm + k] = (_Float16)(soma[r * Dm + k] * inv);
}

// ---------------- GRU tile unit (inlined; keeps LDS/global addrspace chains) ----------------
__device__ __forceinline__ void gru_unit(
    const _Float16 (&Ash)[Bsz * K2D], const _Float16* __restrict__ Wc,
    int mt, int d, int hsel, int l15,
    float bir, float bhr, float biz, float bhz, float bin, float bhn,
    int t, float* __restrict__ p_f, float out[8]) {
  const _Float16* ar = &Ash[(mt * 16 + l15) * K2D];
  const _Float16* wr = Wc + (long)d * K2D;
  const _Float16* wz = Wc + (long)(Dm + d) * K2D;
  const _Float16* wn = Wc + (long)(2 * Dm + d) * K2D;
  v8f accr = {}, accz = {}, accni = {}, accnh = {};
  // input half: K = 0..383 (n-gate -> accni)
  for (int kc = 0; kc < Dm; kc += 32) {
    __builtin_prefetch(wr + kc + 64, 0, 1);
    v16h a = frag_h(ar + kc, hsel);
    accr  = wmma_acc(a, frag_h(wr + kc, hsel), accr);
    accz  = wmma_acc(a, frag_h(wz + kc, hsel), accz);
    accni = wmma_acc(a, frag_h(wn + kc, hsel), accni);
  }
  // hidden half: K = 384..767 (n-gate -> accnh)
  for (int kc = Dm; kc < K2D; kc += 32) {
    v16h a = frag_h(ar + kc, hsel);
    accr  = wmma_acc(a, frag_h(wr + kc, hsel), accr);
    accz  = wmma_acc(a, frag_h(wz + kc, hsel), accz);
    accnh = wmma_acc(a, frag_h(wn + kc, hsel), accnh);
  }
#pragma unroll
  for (int s = 0; s < 8; ++s) {
    int b = mt * 16 + s + 8 * hsel;  // C/D layout: VGPR s -> M = s + 8*hsel
    float r = sigmf(accr[s] + bir + bhr);
    float z = sigmf(accz[s] + biz + bhz);
    float nn = tanhf(accni[s] + bin + r * (accnh[s] + bhn));
    float hold = (float)Ash[b * K2D + Dm + d];
    float hnew = (1.f - z) * nn + z * hold;
    p_f[((long)b * Tlen + t) * Dm + d] = hnew;
    out[s] = hnew;
  }
}

// ---------------- K1: persistent fast GRU (1 block, 32 waves) ----------------
__global__ __launch_bounds__(1024) void k_fast_gru(
    const int* x, const float* h0, const float* soma, const _Float16* Wc,
    const float* bih, const float* bhh, float* p_f, float* hf_out) {
  __shared__ _Float16 A[Bsz * K2D];  // [b][ emb(0..383) | h(384..767) ]  48KB
  __shared__ int toks[Bsz];
  int tid = threadIdx.x;
  int wave = tid >> 5, lane = tid & 31, hsel = lane >> 4, l15 = lane & 15;

  // unit 0: u = wave; unit 1 (waves 0..15 only): u = wave + 32
  int mt0 = wave & 1, d0 = (wave >> 1) * 16 + l15;
  bool has1 = (wave < 16);
  int u1 = wave + 32;
  int mt1 = u1 & 1, d1 = (u1 >> 1) * 16 + l15;
  float bir0 = bih[d0],          bhr0 = bhh[d0];
  float biz0 = bih[Dm + d0],     bhz0 = bhh[Dm + d0];
  float bin0 = bih[2 * Dm + d0], bhn0 = bhh[2 * Dm + d0];
  float bir1 = 0.f, bhr1 = 0.f, biz1 = 0.f, bhz1 = 0.f, bin1 = 0.f, bhn1 = 0.f;
  if (has1) {
    bir1 = bih[d1];          bhr1 = bhh[d1];
    biz1 = bih[Dm + d1];     bhz1 = bhh[Dm + d1];
    bin1 = bih[2 * Dm + d1]; bhn1 = bhh[2 * Dm + d1];
  }

  for (int i = tid; i < Bsz * Dm; i += 1024) {
    int b = i / Dm, d = i % Dm;
    A[b * K2D + Dm + d] = (_Float16)h0[i];
  }
  __syncthreads();

  for (int t = 0; t < Tlen; ++t) {
    if (tid < Bsz) toks[tid] = x[tid * Tlen + t];
    __syncthreads();
    // embedding gather into A[:, 0:384]
    for (int i = tid; i < Bsz * Dm; i += 1024) {
      int b = i / Dm, d = i % Dm;
      A[b * K2D + d] = (_Float16)soma[(long)toks[b] * Dm + d];
    }
    __syncthreads();

    float hn0[8], hn1[8];
    gru_unit(A, Wc, mt0, d0, hsel, l15, bir0, bhr0, biz0, bhz0, bin0, bhn0, t, p_f, hn0);
    if (has1)
      gru_unit(A, Wc, mt1, d1, hsel, l15, bir1, bhr1, biz1, bhz1, bin1, bhn1, t, p_f, hn1);
    __syncthreads();
#pragma unroll
    for (int s = 0; s < 8; ++s) {
      int b = mt0 * 16 + s + 8 * hsel;
      A[b * K2D + Dm + d0] = (_Float16)hn0[s];
    }
    if (has1) {
#pragma unroll
      for (int s = 0; s < 8; ++s) {
        int b = mt1 * 16 + s + 8 * hsel;
        A[b * K2D + Dm + d1] = (_Float16)hn1[s];
      }
    }
    __syncthreads();
  }
  for (int i = tid; i < Bsz * Dm; i += 1024) {
    int b = i / Dm, d = i % Dm;
    hf_out[i] = (float)A[b * K2D + Dm + d];
  }
}

// ---------------- K2: volition head, avg_vol, depth ----------------
__global__ void k_volition(const float* p_f, const float* vg_w, const float* vg_b,
                           float* last_ws, float* ctl, float* out_avg, float* out_depth) {
  __shared__ float vol[32];
  int tid = threadIdx.x;
  for (int i = tid; i < Bsz * Dm; i += 256)
    last_ws[i] = p_f[((long)(i / Dm) * Tlen + (Tlen - 1)) * Dm + (i % Dm)];
  __syncthreads();
  if (tid < 32) {
    float s = vg_b[0];
    for (int k = 0; k < Dm; ++k) s += last_ws[tid * Dm + k] * vg_w[k];
    vol[tid] = sigmf(s);
  }
  __syncthreads();
  if (tid == 0) {
    float a = 0.f;
    for (int b = 0; b < 32; ++b) a += vol[b];
    a /= 32.f;
    int depth = (a > 0.25f) ? (int)(1.f + a * 9.f) : 0;
    ctl[0] = a; ctl[1] = (float)depth;
    *out_avg = a; *out_depth = (float)depth;
  }
}

// ---------------- K3: branch simulation (1 block, 32 waves) ----------------
__global__ __launch_bounds__(1024) void k_branches(
    const float* ctl, const float* last, const _Float16* TrH,
    const float* tr_b, const float* tr_g, const float* tr_beta,
    const float* vh_w, const float* vh_b,
    float* sims, float* vvals, float* best, float* v_score_out, float* bdiff_out) {
  __shared__ float S[Bsz * Dm];  // 48KB
  __shared__ float red[32];
  __shared__ int besti[32];
  int tid = threadIdx.x, wave = tid >> 5, lane = tid & 31, hsel = lane >> 4, l15 = lane & 15;
  int depth = (int)ctl[1];
  if (depth <= 0) {
    for (int i = tid; i < Bsz * Dm; i += 1024) best[i] = last[i];
    if (tid < 32) v_score_out[tid] = 0.f;
    if (tid == 0) *bdiff_out = 0.f;
    return;
  }
  // wave w < 24 owns N-tile w (cols w*16..w*16+15) for BOTH M-tiles; B fragment shared.
  int dcol = wave * 16 + l15;
  const _Float16* wrow = TrH + (long)dcol * Dm;
  float bwv = (wave < 24) ? tr_b[dcol] : 0.f;
  for (int br = 0; br < 3; ++br) {
    for (int i = tid; i < Bsz * Dm; i += 1024) S[i] = last[i];
    __syncthreads();
    float vacc = 0.f;
    for (int st = 0; st < depth; ++st) {
      if (br > 0) {
        for (int i = tid; i < Bsz * Dm; i += 1024)
          S[i] += 0.05f * gnoise((unsigned)(br * 10 + st) * 16384u + (unsigned)i);
        __syncthreads();
      }
      float st0[8], st1[8];
      if (wave < 24) {
        const float* a0 = &S[l15 * Dm];
        const float* a1 = &S[(16 + l15) * Dm];
        v8f acc0 = {}, acc1 = {};
        for (int kc = 0; kc < Dm; kc += 32) {
          v16h b = frag_h(wrow + kc, hsel);
          acc0 = wmma_acc(frag_f(a0 + kc, hsel), b, acc0);
          acc1 = wmma_acc(frag_f(a1 + kc, hsel), b, acc1);
        }
#pragma unroll
        for (int sl = 0; sl < 8; ++sl) {
          float y0 = acc0[sl] + bwv, y1 = acc1[sl] + bwv;
          st0[sl] = y0 > 0.f ? y0 : (expf(y0) - 1.f);  // elu
          st1[sl] = y1 > 0.f ? y1 : (expf(y1) - 1.f);
        }
      }
      __syncthreads();
      if (wave < 24) {
#pragma unroll
        for (int sl = 0; sl < 8; ++sl) {
          int m = sl + 8 * hsel;
          S[m * Dm + dcol] = st0[sl];
          S[(16 + m) * Dm + dcol] = st1[sl];
        }
      }
      __syncthreads();
      // LayerNorm per row (wave == row)
      {
        int row = wave;
        float m = 0.f;
        for (int k = lane; k < Dm; k += 32) m += S[row * Dm + k];
        for (int o = 16; o > 0; o >>= 1) m += __shfl_xor(m, o, 32);
        m /= (float)Dm;
        float v = 0.f;
        for (int k = lane; k < Dm; k += 32) { float dd = S[row * Dm + k] - m; v += dd * dd; }
        for (int o = 16; o > 0; o >>= 1) v += __shfl_xor(v, o, 32);
        v /= (float)Dm;
        float inv = rsqrtf(v + 1e-5f);
        for (int k = lane; k < Dm; k += 32)
          S[row * Dm + k] = (S[row * Dm + k] - m) * inv * tr_g[k] + tr_beta[k];
      }
      __syncthreads();
      // value head accumulate
      {
        int row = wave;
        float dp = 0.f;
        for (int k = lane; k < Dm; k += 32) dp += S[row * Dm + k] * vh_w[k];
        for (int o = 16; o > 0; o >>= 1) dp += __shfl_xor(dp, o, 32);
        vacc += tanhf(dp + vh_b[0]);
      }
      __syncthreads();
    }
    for (int i = tid; i < Bsz * Dm; i += 1024) sims[(long)br * Bsz * Dm + i] = S[i];
    if (lane == 0) vvals[br * Bsz + wave] = vacc / (float)depth;
    __syncthreads();
  }
  if (tid < 32) {
    float v0 = vvals[tid], v1 = vvals[32 + tid], v2 = vvals[64 + tid];
    int bi = 0; float bv = v0;
    if (v1 > bv) { bv = v1; bi = 1; }
    if (v2 > bv) { bv = v2; bi = 2; }
    besti[tid] = bi;
    v_score_out[tid] = bv;
  }
  __syncthreads();
  for (int i = tid; i < Bsz * Dm; i += 1024) {
    int b = i / Dm;
    best[i] = sims[(long)besti[b] * Bsz * Dm + i];
  }
  float part = 0.f;
  for (int i = tid; i < Bsz * Dm; i += 1024) {
    float dd = sims[i] - sims[Bsz * Dm + i];
    part += dd * dd;
  }
  for (int o = 16; o > 0; o >>= 1) part += __shfl_xor(part, o, 32);
  if (lane == 0) red[wave] = part;
  __syncthreads();
  if (tid == 0) {
    float s = 0.f;
    for (int w = 0; w < 32; ++w) s += red[w];
    *bdiff_out = s / (float)(Bsz * Dm);
  }
}

// ---------------- K4: sup & sub single-step GRUs (2 blocks) ----------------
__global__ __launch_bounds__(1024) void k_supsub(
    const float* best, const float* hsup0, const float* hsub0,
    const _Float16* WcS, const _Float16* WcB,
    const float* s_bih, const float* s_bhh, const float* b_bih, const float* b_bhh,
    float* p_sup, float* p_sub, float* hsup_out, float* hsub_out) {
  int g = blockIdx.x;
  const _Float16* Wc = g ? WcB : WcS;
  const float* bih = g ? b_bih : s_bih;
  const float* bhh = g ? b_bhh : s_bhh;
  const float* h0g = g ? hsub0 : hsup0;
  float* pout = g ? p_sub : p_sup;
  float* hout = g ? hsub_out : hsup_out;
  __shared__ _Float16 A[Bsz * K2D];  // 48KB
  int tid = threadIdx.x, wave = tid >> 5, lane = tid & 31, hsel = lane >> 4, l15 = lane & 15;
  for (int i = tid; i < Bsz * Dm; i += 1024) {
    int b = i / Dm, d = i % Dm;
    A[b * K2D + d] = (_Float16)best[i];
    A[b * K2D + Dm + d] = (_Float16)h0g[i];
  }
  __syncthreads();
  for (int u = wave; u < 48; u += 32) {
    int mt = u & 1, dj = u >> 1, d = dj * 16 + l15;
    const _Float16* ar = &A[(mt * 16 + l15) * K2D];
    const _Float16* wr = Wc + (long)d * K2D;
    const _Float16* wz = Wc + (long)(Dm + d) * K2D;
    const _Float16* wn = Wc + (long)(2 * Dm + d) * K2D;
    v8f accr = {}, accz = {}, accni = {}, accnh = {};
    for (int kc = 0; kc < Dm; kc += 32) {
      v16h a = frag_h(ar + kc, hsel);
      accr  = wmma_acc(a, frag_h(wr + kc, hsel), accr);
      accz  = wmma_acc(a, frag_h(wz + kc, hsel), accz);
      accni = wmma_acc(a, frag_h(wn + kc, hsel), accni);
    }
    for (int kc = Dm; kc < K2D; kc += 32) {
      v16h a = frag_h(ar + kc, hsel);
      accr  = wmma_acc(a, frag_h(wr + kc, hsel), accr);
      accz  = wmma_acc(a, frag_h(wz + kc, hsel), accz);
      accnh = wmma_acc(a, frag_h(wn + kc, hsel), accnh);
    }
    float bir = bih[d], bhr = bhh[d];
    float biz = bih[Dm + d], bhz = bhh[Dm + d];
    float bin = bih[2 * Dm + d], bhn = bhh[2 * Dm + d];
#pragma unroll
    for (int s = 0; s < 8; ++s) {
      int b = mt * 16 + s + 8 * hsel;
      float r = sigmf(accr[s] + bir + bhr);
      float z = sigmf(accz[s] + biz + bhz);
      float nn = tanhf(accni[s] + bin + r * (accnh[s] + bhn));
      float hold = (float)A[b * K2D + Dm + d];
      float hnew = (1.f - z) * nn + z * hold;
      pout[b * Dm + d] = hnew;
      hout[b * Dm + d] = hnew;
    }
  }
}

// ---------------- K5: negotiated = gelu(LN(cat(p_sup,p_sub) @ ws_w^T + b)) ----------------
__global__ __launch_bounds__(1024) void k_negotiate(
    const float* p_sup, const float* p_sub, const _Float16* WsH,
    const float* ws_b, const float* ws_g, const float* ws_beta, float* neg) {
  __shared__ float Y[Bsz * Dm];  // 48KB
  int tid = threadIdx.x, wave = tid >> 5, lane = tid & 31, hsel = lane >> 4, l15 = lane & 15;
  if (wave < 24) {
    int d = wave * 16 + l15;
    const _Float16* w = WsH + (long)d * K2D;
    v8f acc0 = {}, acc1 = {};
    for (int kc = 0; kc < Dm; kc += 32) {       // p_sup half
      v16h b = frag_h(w + kc, hsel);
      acc0 = wmma_acc(frag_f(p_sup + l15 * Dm + kc, hsel), b, acc0);
      acc1 = wmma_acc(frag_f(p_sup + (16 + l15) * Dm + kc, hsel), b, acc1);
    }
    for (int kc = 0; kc < Dm; kc += 32) {       // p_sub half
      v16h b = frag_h(w + Dm + kc, hsel);
      acc0 = wmma_acc(frag_f(p_sub + l15 * Dm + kc, hsel), b, acc0);
      acc1 = wmma_acc(frag_f(p_sub + (16 + l15) * Dm + kc, hsel), b, acc1);
    }
    float bb = ws_b[d];
#pragma unroll
    for (int sl = 0; sl < 8; ++sl) {
      int m = sl + 8 * hsel;
      Y[m * Dm + d] = acc0[sl] + bb;
      Y[(16 + m) * Dm + d] = acc1[sl] + bb;
    }
  }
  __syncthreads();
  {
    int row = wave;
    float m = 0.f;
    for (int k = lane; k < Dm; k += 32) m += Y[row * Dm + k];
    for (int o = 16; o > 0; o >>= 1) m += __shfl_xor(m, o, 32);
    m /= (float)Dm;
    float v = 0.f;
    for (int k = lane; k < Dm; k += 32) { float dd = Y[row * Dm + k] - m; v += dd * dd; }
    for (int o = 16; o > 0; o >>= 1) v += __shfl_xor(v, o, 32);
    v /= (float)Dm;
    float inv = rsqrtf(v + 1e-5f);
    for (int k = lane; k < Dm; k += 32) {
      float val = (Y[row * Dm + k] - m) * inv * ws_g[k] + ws_beta[k];
      neg[row * Dm + k] = geluf(val);
    }
  }
}

// ---------------- K6: fused pfc + LN + gelu + LN + logits (2048 blocks) ----------------
__global__ __launch_bounds__(256) void k_thought_logits(
    const float* p_f, const float* neg, const _Float16* PfcH,
    const float* pfc_b, const float* pfc_g, const float* pfc_beta,
    const float* on_g, const float* on_beta,
    const _Float16* swn, float* logits) {
  __shared__ float TH[16 * Dm];  // 24KB
  long row0 = (long)blockIdx.x * 16;
  int bidx = (int)(row0 / Tlen);
  int tid = threadIdx.x, wave = tid >> 5, lane = tid & 31, hsel = lane >> 4, l15 = lane & 15;
  const float* nrow = neg + (long)bidx * Dm;
  // pfc GEMM: 16 x 384 (K = 768); wave owns 3 N-tiles: cols d0, d0+128, d0+256.
  {
    int d0 = wave * 16 + l15;
    const _Float16* w0 = PfcH + (long)d0 * K2D;
    const _Float16* w1 = PfcH + (long)(d0 + 128) * K2D;
    const _Float16* w2 = PfcH + (long)(d0 + 256) * K2D;
    const float* arow = p_f + (row0 + l15) * Dm;
    v8f a0 = {}, a1 = {}, a2 = {};
    for (int kc = 0; kc < Dm; kc += 32) {       // K half from p_f
      __builtin_prefetch(w0 + kc + 64, 0, 1);
      v16h a = frag_f(arow + kc, hsel);
      a0 = wmma_acc(a, frag_h(w0 + kc, hsel), a0);
      a1 = wmma_acc(a, frag_h(w1 + kc, hsel), a1);
      a2 = wmma_acc(a, frag_h(w2 + kc, hsel), a2);
    }
    for (int kc = 0; kc < Dm; kc += 32) {       // K half from negotiated (broadcast row)
      v16h a = frag_f(nrow + kc, hsel);
      a0 = wmma_acc(a, frag_h(w0 + Dm + kc, hsel), a0);
      a1 = wmma_acc(a, frag_h(w1 + Dm + kc, hsel), a1);
      a2 = wmma_acc(a, frag_h(w2 + Dm + kc, hsel), a2);
    }
    float b0 = pfc_b[d0], b1 = pfc_b[d0 + 128], b2 = pfc_b[d0 + 256];
#pragma unroll
    for (int sl = 0; sl < 8; ++sl) {
      int m = sl + 8 * hsel;
      TH[m * Dm + d0]       = a0[sl] + b0;
      TH[m * Dm + d0 + 128] = a1[sl] + b1;
      TH[m * Dm + d0 + 256] = a2[sl] + b2;
    }
  }
  __syncthreads();
  // LN1 + gelu
  for (int r = wave; r < 16; r += 8) {
    float m = 0.f;
    for (int k = lane; k < Dm; k += 32) m += TH[r * Dm + k];
    for (int o = 16; o > 0; o >>= 1) m += __shfl_xor(m, o, 32);
    m /= (float)Dm;
    float v = 0.f;
    for (int k = lane; k < Dm; k += 32) { float dd = TH[r * Dm + k] - m; v += dd * dd; }
    for (int o = 16; o > 0; o >>= 1) v += __shfl_xor(v, o, 32);
    v /= (float)Dm;
    float inv = rsqrtf(v + 1e-5f);
    for (int k = lane; k < Dm; k += 32) {
      float val = (TH[r * Dm + k] - m) * inv * pfc_g[k] + pfc_beta[k];
      TH[r * Dm + k] = geluf(val);
    }
  }
  __syncthreads();
  // LN2 (on_g / on_beta)
  for (int r = wave; r < 16; r += 8) {
    float m = 0.f;
    for (int k = lane; k < Dm; k += 32) m += TH[r * Dm + k];
    for (int o = 16; o > 0; o >>= 1) m += __shfl_xor(m, o, 32);
    m /= (float)Dm;
    float v = 0.f;
    for (int k = lane; k < Dm; k += 32) { float dd = TH[r * Dm + k] - m; v += dd * dd; }
    for (int o = 16; o > 0; o >>= 1) v += __shfl_xor(v, o, 32);
    v /= (float)Dm;
    float inv = rsqrtf(v + 1e-5f);
    for (int k = lane; k < Dm; k += 32)
      TH[r * Dm + k] = (TH[r * Dm + k] - m) * inv * on_g[k] + on_beta[k];
  }
  __syncthreads();
  // logits GEMM: 16 x 256 (K = 384); wave owns 2 N-tiles: cols n0, n0+128.
  {
    int n0 = wave * 16 + l15;
    const _Float16* w0 = swn + (long)n0 * Dm;
    const _Float16* w1 = swn + (long)(n0 + 128) * Dm;
    const float* arow = &TH[l15 * Dm];
    v8f l0 = {}, l1 = {};
    for (int kc = 0; kc < Dm; kc += 32) {
      v16h a = frag_f(arow + kc, hsel);
      l0 = wmma_acc(a, frag_h(w0 + kc, hsel), l0);
      l1 = wmma_acc(a, frag_h(w1 + kc, hsel), l1);
    }
#pragma unroll
    for (int sl = 0; sl < 8; ++sl) {
      int m = sl + 8 * hsel;
      logits[(row0 + m) * VOC + n0]       = l0[sl] * GAINF;
      logits[(row0 + m) * VOC + n0 + 128] = l1[sl] * GAINF;
    }
  }
}

// ---------------- host-side orchestration ----------------
extern "C" void kernel_launch(void* const* d_in, const int* in_sizes, int n_in,
                              void* d_out, int out_size, void* d_ws, size_t ws_size,
                              hipStream_t stream) {
  const int*   x        = (const int*)d_in[0];
  const float* h_f      = (const float*)d_in[1];
  const float* h_super  = (const float*)d_in[2];
  const float* h_sub    = (const float*)d_in[3];
  const float* soma_w   = (const float*)d_in[4];
  const float* fast_wih = (const float*)d_in[5];
  const float* fast_whh = (const float*)d_in[6];
  const float* fast_bih = (const float*)d_in[7];
  const float* fast_bhh = (const float*)d_in[8];
  const float* sup_wih  = (const float*)d_in[9];
  const float* sup_whh  = (const float*)d_in[10];
  const float* sup_bih  = (const float*)d_in[11];
  const float* sup_bhh  = (const float*)d_in[12];
  const float* sub_wih  = (const float*)d_in[13];
  const float* sub_whh  = (const float*)d_in[14];
  const float* sub_bih  = (const float*)d_in[15];
  const float* sub_bhh  = (const float*)d_in[16];
  const float* ws_w     = (const float*)d_in[17];
  const float* ws_b     = (const float*)d_in[18];
  const float* ws_g     = (const float*)d_in[19];
  const float* ws_beta  = (const float*)d_in[20];
  const float* vg_w     = (const float*)d_in[21];
  const float* vg_b     = (const float*)d_in[22];
  const float* tr_w     = (const float*)d_in[23];
  const float* tr_b     = (const float*)d_in[24];
  const float* tr_g     = (const float*)d_in[25];
  const float* tr_beta  = (const float*)d_in[26];
  const float* vh_w     = (const float*)d_in[27];
  const float* vh_b     = (const float*)d_in[28];
  const float* pfc_w    = (const float*)d_in[29];
  const float* pfc_b    = (const float*)d_in[30];
  const float* pfc_g    = (const float*)d_in[31];
  const float* pfc_beta = (const float*)d_in[32];
  const float* on_g     = (const float*)d_in[33];
  const float* on_beta  = (const float*)d_in[34];

  float* out = (float*)d_out;
  float* out_logits = out;                       // 32*1024*256
  float* out_hf     = out + 8388608;             // 12288
  float* out_hsup   = out + 8400896;             // 12288
  float* out_hsub   = out + 8413184;             // 12288
  float* out_vsc    = out + 8425472;             // 32
  float* out_avg    = out + 8425504;             // 1
  float* out_depth  = out + 8425505;             // 1
  float* out_bdiff  = out + 8425506;             // 1

  float* ws = (float*)d_ws;   // offsets in floats
  _Float16* WcF  = (_Float16*)(ws + 0);          // 1152x768 f16
  _Float16* WcS  = (_Float16*)(ws + 442368);
  _Float16* WcB  = (_Float16*)(ws + 884736);
  _Float16* TrH  = (_Float16*)(ws + 1327104);    // 384x384 f16
  _Float16* WsH  = (_Float16*)(ws + 1400832);    // 384x768 f16
  _Float16* PfcH = (_Float16*)(ws + 1548288);    // 384x768 f16
  _Float16* SwN  = (_Float16*)(ws + 1695744);    // 256x384 f16 (normalized soma)
  float* p_f     = ws + 1744896;                 // 32*1024*384
  float* last_ws = ws + 14327808;                // 32*384
  float* ctl     = ws + 14340096;                // 16
  float* sims    = ws + 14340112;                // 3*32*384
  float* vvals   = ws + 14376976;                // 3*32
  float* best    = ws + 14377072;                // 32*384
  float* p_sup   = ws + 14389360;                // 32*384
  float* p_sub   = ws + 14401648;                // 32*384
  float* negb    = ws + 14413936;                // 32*384

  k_prep<<<256, 256, 0, stream>>>(fast_wih, fast_whh, sup_wih, sup_whh, sub_wih, sub_whh,
                                  tr_w, ws_w, pfc_w, WcF, WcS, WcB, TrH, WsH, PfcH);
  k_norm_soma<<<1, 256, 0, stream>>>(soma_w, SwN);
  k_fast_gru<<<1, 1024, 0, stream>>>(x, h_f, soma_w, WcF, fast_bih, fast_bhh, p_f, out_hf);
  k_volition<<<1, 256, 0, stream>>>(p_f, vg_w, vg_b, last_ws, ctl, out_avg, out_depth);
  k_branches<<<1, 1024, 0, stream>>>(ctl, last_ws, TrH, tr_b, tr_g, tr_beta, vh_w, vh_b,
                                     sims, vvals, best, out_vsc, out_bdiff);
  k_supsub<<<2, 1024, 0, stream>>>(best, h_super, h_sub, WcS, WcB,
                                   sup_bih, sup_bhh, sub_bih, sub_bhh,
                                   p_sup, p_sub, out_hsup, out_hsub);
  k_negotiate<<<1, 1024, 0, stream>>>(p_sup, p_sub, WsH, ws_b, ws_g, ws_beta, negb);
  k_thought_logits<<<2048, 256, 0, stream>>>(p_f, negb, PfcH, pfc_b, pfc_g, pfc_beta,
                                             on_g, on_beta, SwN, out_logits);
}
